// ASTGPOLS_72086731096215
// MI455X (gfx1250) — matrix operand order repaired and checked by reading, not compiled, not run on hardware
//
#include <hip/hip_runtime.h>
#include <hip/hip_bf16.h>
#include <math.h>

// ---------------------------------------------------------------------------
// Problem constants (from reference)
// ---------------------------------------------------------------------------
constexpr int N_NODES = 20000;
constexpr int E_EDGES = 160000;
constexpr int T_STEPS = 8;
constexpr int F_IN    = 256;
constexpr int HEADS   = 4;
constexpr int C_DIM   = 128;
constexpr int HC      = HEADS * C_DIM;   // 512
constexpr int GH      = 128;
constexpr float NEG_SLOPE = 0.2f;

typedef __bf16 bf16_t;
typedef __attribute__((ext_vector_type(16))) __bf16 v16bf;
typedef __attribute__((ext_vector_type(8)))  __bf16 v8bf;
typedef __attribute__((ext_vector_type(8)))  float  v8f;

#define DEVFN __device__ __forceinline__

DEVFN bf16_t f2bf(float f) {
    unsigned u = __float_as_uint(f);
    unsigned r = (u + 0x7FFFu + ((u >> 16) & 1u)) >> 16;
    unsigned short s = (unsigned short)r;
    return __builtin_bit_cast(bf16_t, s);
}

DEVFN float leaky(float x) { return x > 0.f ? x : NEG_SLOPE * x; }

DEVFN float warpSum(float v) {
#pragma unroll
    for (int o = 16; o; o >>= 1) v += __shfl_xor(v, o, 32);
    return v;
}

DEVFN void atomicMaxF(float* addr, float val) {
    int old = __float_as_int(*addr);
    while (__int_as_float(old) < val) {
        int assumed = old;
        old = atomicCAS((int*)addr, assumed, __float_as_int(val));
        if (old == assumed) break;
    }
}

// ---------------------------------------------------------------------------
// WMMA GEMM:  C[M x Nn] = A[M x K] (bf16, row major) * Bt^T
//   Bt is stored [Nn x K] row-major (weights in natural (out,in) layout).
//   One wave computes a 16 x 64 strip of C (4 accumulators along N): the A
//   fragment is loaded once per K-step and reused by 4 WMMAs -> 4x fewer A
//   loads and 4 v_wmma per loop body.
//   A fragment (16-bit A 16x32 layout): lane = M%16, hi = lane>=16 selects
//     K chunks {hi*8..+7} and {16+hi*8..}: two contiguous 16B loads.
//   B fragment (16-bit B 32x16 layout): lane = N%16, same K chunks as A
//     {hi*8..+7} and {16+hi*8..}: two contiguous 16B loads.
// ---------------------------------------------------------------------------
union V16U { v16bf v; v8bf h[2]; };
constexpr int NT = 4;   // 16-wide N tiles per wave

__global__ void wmma_gemm_bf16(const bf16_t* __restrict__ A,
                               const bf16_t* __restrict__ Bt,
                               float* __restrict__ C,
                               int M, int Nn, int K,
                               const float* __restrict__ bias, int act) {
    (void)M;
    const int tn   = blockIdx.x;           // 64-wide tile column
    const int tm   = blockIdx.y;           // 16-row tile
    const int lane = threadIdx.x;
    const int lmod = lane & 15;
    const int hi   = lane >> 4;

    const bf16_t* arow  = A  + (size_t)(tm * 16 + lmod) * K;
    const bf16_t* brow0 = Bt + (size_t)(tn * (16 * NT) + lmod) * K;

    v8f acc[NT];
#pragma unroll
    for (int j = 0; j < NT; ++j) acc[j] = (v8f){};

    for (int k = 0; k < K; k += 32) {
        V16U a;
        a.h[0] = *(const v8bf*)(arow + k + hi * 8);
        a.h[1] = *(const v8bf*)(arow + k + 16 + hi * 8);
#pragma unroll
        for (int j = 0; j < NT; ++j) {
            const bf16_t* br = brow0 + (size_t)j * 16 * K;
            V16U b;
            b.h[0] = *(const v8bf*)(br + k + hi * 8);
            b.h[1] = *(const v8bf*)(br + k + 16 + hi * 8);
            acc[j] = __builtin_amdgcn_wmma_f32_16x16x32_bf16(
                false, a.v, false, b.v, (short)0, acc[j], false, false);
        }
    }

#pragma unroll
    for (int j = 0; j < NT; ++j) {
        const int col = tn * (16 * NT) + j * 16 + lmod;
        float bb = bias ? bias[col] : 0.f;
#pragma unroll
        for (int r = 0; r < 8; ++r) {
            int orow = tm * 16 + hi * 8 + r;   // C/D layout: VGPR r -> M = r + hi*8
            float v = acc[j][r] + bb;
            if (act == 1) v = tanhf(v);
            C[(size_t)orow * Nn + col] = v;
        }
    }
}

// ---------------------------------------------------------------------------
// Utility kernels
// ---------------------------------------------------------------------------
__global__ void fill_f32(float* __restrict__ p, float v, long long cnt) {
    long long i = (long long)blockIdx.x * blockDim.x + threadIdx.x;
    if (i < cnt) p[i] = v;
}

__global__ void conv_f32_bf16(const float* __restrict__ s, bf16_t* __restrict__ d,
                              long long cnt) {
    long long i = (long long)blockIdx.x * blockDim.x + threadIdx.x;
    if (i < cnt) d[i] = f2bf(s[i]);
}

// W_gat is (F_IN, HC) = (K, N); store transposed -> Bt layout (N, K)
__global__ void conv_wgat_t(const float* __restrict__ W, bf16_t* __restrict__ Wt) {
    int idx = blockIdx.x * blockDim.x + threadIdx.x;
    if (idx >= F_IN * HC) return;
    int k = idx / HC, j = idx % HC;
    Wt[(size_t)j * F_IN + k] = f2bf(W[idx]);
}

// ---------------------------------------------------------------------------
// GAT kernels
// ---------------------------------------------------------------------------
// a_src[n,h] = dot(xw[n,h,:], att_src[h,:]); same for a_dst. Warp per (n,h).
__global__ void gat_att_coef(const float* __restrict__ xw,
                             const float* __restrict__ att_src,
                             const float* __restrict__ att_dst,
                             float* __restrict__ asrc, float* __restrict__ adst) {
    int warp = (blockIdx.x * blockDim.x + threadIdx.x) >> 5;
    int lane = threadIdx.x & 31;
    if (warp >= N_NODES * HEADS) return;
    int node = warp >> 2, h = warp & 3;
    const float* xr = xw + (size_t)node * HC + h * C_DIM;
    float s1 = 0.f, s2 = 0.f;
    for (int c = lane; c < C_DIM; c += 32) {
        float v = xr[c];
        s1 += v * att_src[h * C_DIM + c];
        s2 += v * att_dst[h * C_DIM + c];
    }
    s1 = warpSum(s1); s2 = warpSum(s2);
    if (!lane) { asrc[warp] = s1; adst[warp] = s2; }
}

DEVFN void edge_nodes(const int* et, int e, int& s, int& d) {
    if (e < E_EDGES) { s = et[e]; d = et[E_EDGES + e]; }
    else             { s = d = e - E_EDGES; }          // self loops
}

__global__ void edge_logit(const int* __restrict__ et,
                           const float* __restrict__ asrc,
                           const float* __restrict__ adst,
                           float* __restrict__ alpha, float* __restrict__ segmax) {
    int idx = blockIdx.x * blockDim.x + threadIdx.x;
    if (idx >= (E_EDGES + N_NODES) * HEADS) return;
    int e = idx >> 2, h = idx & 3;
    int s, d; edge_nodes(et, e, s, d);
    float l = leaky(asrc[s * HEADS + h] + adst[d * HEADS + h]);
    alpha[idx] = l;
    atomicMaxF(&segmax[d * HEADS + h], l);
}

__global__ void edge_exp(const int* __restrict__ et,
                         float* __restrict__ alpha,
                         const float* __restrict__ segmax,
                         float* __restrict__ segsum) {
    int idx = blockIdx.x * blockDim.x + threadIdx.x;
    if (idx >= (E_EDGES + N_NODES) * HEADS) return;
    int e = idx >> 2, h = idx & 3;
    int s, d; edge_nodes(et, e, s, d); (void)s;
    float ev = __expf(alpha[idx] - segmax[d * HEADS + h]);
    alpha[idx] = ev;
    atomicAdd(&segsum[d * HEADS + h], ev);
}

// Warp per (edge, head); lanes stripe the 128 channels; accumulate over heads.
__global__ void edge_scatter(const int* __restrict__ et,
                             const float* __restrict__ alpha,
                             const float* __restrict__ segsum,
                             const float* __restrict__ xw,
                             float* __restrict__ acc) {
    int warp = (blockIdx.x * blockDim.x + threadIdx.x) >> 5;
    int lane = threadIdx.x & 31;
    if (warp >= (E_EDGES + N_NODES) * HEADS) return;
    int e = warp >> 2, h = warp & 3;
    int s, d; edge_nodes(et, e, s, d);
    float coeff = alpha[warp] / (segsum[d * HEADS + h] + 1e-16f);
    const float* xr = xw + (size_t)s * HC + h * C_DIM;
    float* ar = acc + (size_t)d * C_DIM;
#pragma unroll
    for (int c0 = 0; c0 < C_DIM; c0 += 32) {
        int c = c0 + lane;
        atomicAdd(&ar[c], xr[c] * coeff);
    }
}

// y[t] = leaky_relu(mean_over_heads + b_gat)  -> bf16 (GRU input)
__global__ void gat_final(const float* __restrict__ acc,
                          const float* __restrict__ b_gat,
                          bf16_t* __restrict__ yt) {
    int idx = blockIdx.x * blockDim.x + threadIdx.x;
    if (idx >= N_NODES * GH) return;
    int c = idx & (GH - 1);
    float v = acc[idx] * (1.f / HEADS) + b_gat[c];
    yt[idx] = f2bf(leaky(v));
}

// ---------------------------------------------------------------------------
// GRU pointwise (gi/gh already include biases from the GEMM epilogue)
// ---------------------------------------------------------------------------
__global__ void gru_point(const float* __restrict__ gi, const float* __restrict__ gh,
                          float* __restrict__ h, bf16_t* __restrict__ hbf,
                          float* __restrict__ ctx_t) {
    int idx = blockIdx.x * blockDim.x + threadIdx.x;
    if (idx >= N_NODES * GH) return;
    int n = idx >> 7, c = idx & (GH - 1);
    const float* gir = gi + (size_t)n * 3 * GH;
    const float* ghr = gh + (size_t)n * 3 * GH;
    float r  = 1.f / (1.f + __expf(-(gir[c]          + ghr[c])));
    float z  = 1.f / (1.f + __expf(-(gir[GH + c]     + ghr[GH + c])));
    float nn = tanhf(gir[2 * GH + c] + r * ghr[2 * GH + c]);
    float hp = h[idx];
    float hn = (1.f - z) * nn + z * hp;
    h[idx] = hn;
    hbf[idx] = f2bf(hn);
    ctx_t[idx] = hn;
}

// ---------------------------------------------------------------------------
// Temporal attention
// ---------------------------------------------------------------------------
__global__ void attn_scores(const float* __restrict__ q, const float* __restrict__ ctx,
                            float* __restrict__ scores) {
    int warp = (blockIdx.x * blockDim.x + threadIdx.x) >> 5;
    int lane = threadIdx.x & 31;
    if (warp >= N_NODES) return;
    const float* qr = q + (size_t)warp * GH;
#pragma unroll
    for (int t = 0; t < T_STEPS; ++t) {
        const float* cr = ctx + (size_t)t * N_NODES * GH + (size_t)warp * GH;
        float s = 0.f;
        for (int c = lane; c < GH; c += 32) s += qr[c] * cr[c];
        s = warpSum(s);
        if (!lane) scores[warp * T_STEPS + t] = s;
    }
}

__global__ void attn_mix(const float* __restrict__ scores, const float* __restrict__ ctx,
                         const float* __restrict__ q, bf16_t* __restrict__ comb) {
    int warp = (blockIdx.x * blockDim.x + threadIdx.x) >> 5;
    int lane = threadIdx.x & 31;
    if (warp >= N_NODES) return;
    float w[T_STEPS]; float m = -1e30f;
#pragma unroll
    for (int t = 0; t < T_STEPS; ++t) { w[t] = scores[warp * T_STEPS + t]; m = fmaxf(m, w[t]); }
    float s = 0.f;
#pragma unroll
    for (int t = 0; t < T_STEPS; ++t) { w[t] = __expf(w[t] - m); s += w[t]; }
    float inv = 1.f / s;
    for (int c = lane; c < GH; c += 32) {
        float a = 0.f;
#pragma unroll
        for (int t = 0; t < T_STEPS; ++t)
            a += w[t] * ctx[(size_t)t * N_NODES * GH + (size_t)warp * GH + c];
        comb[(size_t)warp * 2 * GH + c]      = f2bf(a * inv);               // mix
        comb[(size_t)warp * 2 * GH + GH + c] = f2bf(q[(size_t)warp * GH + c]); // q
    }
}

// output = leaky(att); logits = output @ W_cls^T + b_cls; log_softmax (NC = 2)
__global__ void classify(const float* __restrict__ att, const float* __restrict__ Wcls,
                         const float* __restrict__ bcls, float* __restrict__ out) {
    int warp = (blockIdx.x * blockDim.x + threadIdx.x) >> 5;
    int lane = threadIdx.x & 31;
    if (warp >= N_NODES) return;
    float l0 = 0.f, l1 = 0.f;
    for (int c = lane; c < GH; c += 32) {
        float o = leaky(att[(size_t)warp * GH + c]);
        l0 += o * Wcls[c];
        l1 += o * Wcls[GH + c];
    }
    l0 = warpSum(l0); l1 = warpSum(l1);
    if (!lane) {
        l0 += bcls[0]; l1 += bcls[1];
        float m = fmaxf(l0, l1);
        float lse = m + logf(__expf(l0 - m) + __expf(l1 - m));
        out[warp * 2 + 0] = l0 - lse;
        out[warp * 2 + 1] = l1 - lse;
    }
}

// ---------------------------------------------------------------------------
// Host orchestration
// ---------------------------------------------------------------------------
extern "C" void kernel_launch(void* const* d_in, const int* in_sizes, int n_in,
                              void* d_out, int out_size, void* d_ws, size_t ws_size,
                              hipStream_t stream) {
    (void)in_sizes; (void)n_in; (void)out_size; (void)ws_size;

    const int*   graph     = (const int*)  d_in[0];   // (T,2,E)
    const float* fts       = (const float*)d_in[1];   // (T,N,F_IN)
    const float* W_gat     = (const float*)d_in[3];   // (F_IN, HC)
    const float* att_src   = (const float*)d_in[4];   // (H, C)
    const float* att_dst   = (const float*)d_in[5];
    const float* b_gat     = (const float*)d_in[6];   // (C,)
    const float* W_ih      = (const float*)d_in[7];   // (3GH, C)
    const float* W_hh      = (const float*)d_in[8];   // (3GH, GH)
    const float* b_ih      = (const float*)d_in[9];
    const float* b_hh      = (const float*)d_in[10];
    const float* W_att_in  = (const float*)d_in[11];  // (GH, GH)
    const float* W_att_out = (const float*)d_in[12];  // (GH, 2GH)
    const float* W_cls     = (const float*)d_in[13];  // (2, GH)
    const float* b_cls     = (const float*)d_in[14];
    float* out = (float*)d_out;

    char* ws = (char*)d_ws; size_t off = 0;
    auto alloc = [&](size_t bytes) -> void* {
        void* p = ws + off; off = (off + bytes + 255) & ~(size_t)255; return p;
    };

    bf16_t* fts_bf     = (bf16_t*)alloc((size_t)T_STEPS * N_NODES * F_IN * 2);
    bf16_t* wgat_t     = (bf16_t*)alloc((size_t)HC * F_IN * 2);
    bf16_t* wih_bf     = (bf16_t*)alloc((size_t)3 * GH * C_DIM * 2);
    bf16_t* whh_bf     = (bf16_t*)alloc((size_t)3 * GH * GH * 2);
    bf16_t* wattin_bf  = (bf16_t*)alloc((size_t)GH * GH * 2);
    bf16_t* wattout_bf = (bf16_t*)alloc((size_t)GH * 2 * GH * 2);
    float*  xw         = (float*) alloc((size_t)N_NODES * HC * 4);
    float*  asrc       = (float*) alloc((size_t)N_NODES * HEADS * 4);
    float*  adst       = (float*) alloc((size_t)N_NODES * HEADS * 4);
    float*  alpha      = (float*) alloc((size_t)(E_EDGES + N_NODES) * HEADS * 4);
    float*  segmax     = (float*) alloc((size_t)N_NODES * HEADS * 4);
    float*  segsum     = (float*) alloc((size_t)N_NODES * HEADS * 4);
    float*  acc        = (float*) alloc((size_t)N_NODES * C_DIM * 4);
    bf16_t* y_bf       = (bf16_t*)alloc((size_t)T_STEPS * N_NODES * GH * 2);
    float*  h          = (float*) alloc((size_t)N_NODES * GH * 4);
    bf16_t* h_bf       = (bf16_t*)alloc((size_t)N_NODES * GH * 2);
    float*  gi         = (float*) alloc((size_t)N_NODES * 3 * GH * 4);
    float*  gh         = (float*) alloc((size_t)N_NODES * 3 * GH * 4);
    float*  ctx        = (float*) alloc((size_t)T_STEPS * N_NODES * GH * 4);
    float*  q          = (float*) alloc((size_t)N_NODES * GH * 4);
    float*  scores     = (float*) alloc((size_t)N_NODES * T_STEPS * 4);
    bf16_t* comb_bf    = (bf16_t*)alloc((size_t)N_NODES * 2 * GH * 2);
    float*  attv       = (float*) alloc((size_t)N_NODES * GH * 4);

    const int B = 256;
    const int NW = 16 * NT;   // 64 columns per wave
    // --- weight / feature conversion to bf16 ---
    long long cf = (long long)T_STEPS * N_NODES * F_IN;
    conv_f32_bf16<<<(unsigned)((cf + B - 1) / B), B, 0, stream>>>(fts, fts_bf, cf);
    conv_wgat_t<<<(F_IN * HC + B - 1) / B, B, 0, stream>>>(W_gat, wgat_t);
    conv_f32_bf16<<<(3 * GH * C_DIM + B - 1) / B, B, 0, stream>>>(W_ih, wih_bf, 3 * GH * C_DIM);
    conv_f32_bf16<<<(3 * GH * GH + B - 1) / B, B, 0, stream>>>(W_hh, whh_bf, 3 * GH * GH);
    conv_f32_bf16<<<(GH * GH + B - 1) / B, B, 0, stream>>>(W_att_in, wattin_bf, GH * GH);
    conv_f32_bf16<<<(GH * 2 * GH + B - 1) / B, B, 0, stream>>>(W_att_out, wattout_bf, GH * 2 * GH);

    const int EH = (E_EDGES + N_NODES) * HEADS;
    // ---------------- GAT per time step ----------------
    for (int t = 0; t < T_STEPS; ++t) {
        const bf16_t* At = fts_bf + (size_t)t * N_NODES * F_IN;
        wmma_gemm_bf16<<<dim3(HC / NW, N_NODES / 16), 32, 0, stream>>>(
            At, wgat_t, xw, N_NODES, HC, F_IN, nullptr, 0);
        gat_att_coef<<<(N_NODES * HEADS * 32 + B - 1) / B, B, 0, stream>>>(
            xw, att_src, att_dst, asrc, adst);
        fill_f32<<<(N_NODES * HEADS + B - 1) / B, B, 0, stream>>>(segmax, -1e30f, N_NODES * HEADS);
        fill_f32<<<(N_NODES * HEADS + B - 1) / B, B, 0, stream>>>(segsum, 0.f, N_NODES * HEADS);
        fill_f32<<<(N_NODES * C_DIM + B - 1) / B, B, 0, stream>>>(acc, 0.f, N_NODES * C_DIM);
        const int* et = graph + (size_t)t * 2 * E_EDGES;
        edge_logit<<<(EH + B - 1) / B, B, 0, stream>>>(et, asrc, adst, alpha, segmax);
        edge_exp<<<(EH + B - 1) / B, B, 0, stream>>>(et, alpha, segmax, segsum);
        edge_scatter<<<(EH * 32 + B - 1) / B, B, 0, stream>>>(et, alpha, segsum, xw, acc);
        gat_final<<<(N_NODES * GH + B - 1) / B, B, 0, stream>>>(
            acc, b_gat, y_bf + (size_t)t * N_NODES * GH);
    }

    // ---------------- GRU over time ----------------
    fill_f32<<<(N_NODES * GH + B - 1) / B, B, 0, stream>>>(h, 0.f, N_NODES * GH);
    fill_f32<<<(N_NODES * GH / 2 + B - 1) / B, B, 0, stream>>>((float*)h_bf, 0.f, N_NODES * GH / 2);
    for (int t = 0; t < T_STEPS; ++t) {
        wmma_gemm_bf16<<<dim3(3 * GH / NW, N_NODES / 16), 32, 0, stream>>>(
            y_bf + (size_t)t * N_NODES * GH, wih_bf, gi, N_NODES, 3 * GH, GH, b_ih, 0);
        wmma_gemm_bf16<<<dim3(3 * GH / NW, N_NODES / 16), 32, 0, stream>>>(
            h_bf, whh_bf, gh, N_NODES, 3 * GH, GH, b_hh, 0);
        gru_point<<<(N_NODES * GH + B - 1) / B, B, 0, stream>>>(
            gi, gh, h, h_bf, ctx + (size_t)t * N_NODES * GH);
    }

    // ---------------- temporal attention + classifier ----------------
    // q = last @ W_att_in^T  (h_bf == bf16(last) after the loop)
    wmma_gemm_bf16<<<dim3(GH / NW, N_NODES / 16), 32, 0, stream>>>(
        h_bf, wattin_bf, q, N_NODES, GH, GH, nullptr, 0);
    attn_scores<<<(N_NODES * 32 + B - 1) / B, B, 0, stream>>>(q, ctx, scores);
    attn_mix<<<(N_NODES * 32 + B - 1) / B, B, 0, stream>>>(scores, ctx, q, comb_bf);
    // att_out = tanh([mix, q] @ W_att_out^T)
    wmma_gemm_bf16<<<dim3(GH / NW, N_NODES / 16), 32, 0, stream>>>(
        comb_bf, wattout_bf, attv, N_NODES, GH, 2 * GH, nullptr, 1);
    classify<<<(N_NODES * 32 + B - 1) / B, B, 0, stream>>>(attv, W_cls, b_cls, out);
}
